// FAGCNExpertLite_54692113547708
// MI455X (gfx1250) — matrix operand order repaired and checked by reading, not compiled, not run on hardware
//
#include <hip/hip_runtime.h>

#define N_NODES 50000
#define N_EDGES 800000
#define F_INDIM 128
#define H_DIM   64
#define N_LAYERS 4
#define N_GRAPHS 64
#define EPS_F   0.1f

typedef __attribute__((ext_vector_type(16))) __bf16          v16bf;
typedef __attribute__((ext_vector_type(16))) unsigned short  v16u;
typedef __attribute__((ext_vector_type(8)))  float           v8f;

static __device__ __forceinline__ unsigned short f2bf(float f) {
  // round-to-nearest-even fp32 -> bf16
  unsigned int u = __float_as_uint(f);
  u += 0x7fffu + ((u >> 16) & 1u);
  return (unsigned short)(u >> 16);
}

// one-shot weight conversion f32 -> bf16 (tiny: 8K + 4K elements per launch)
__global__ void cvt_bf16_kernel(const float* __restrict__ in,
                                unsigned short* __restrict__ out, int n) {
  int i = blockIdx.x * blockDim.x + threadIdx.x;
  if (i < n) out[i] = f2bf(in[i]);
}

// ---------------------------------------------------------------------------
// WMMA GEMM: out[M,64] = act(A[M,K] @ B[K,64] + bias).
// A: f32, converted to bf16 in-register (RNE). B: pre-converted bf16, loaded
// as 16 strided 16-bit loads per n-tile (one per k row, stride H_DIM).
// f32 accumulate via v_wmma_f32_16x16x32_bf16; one wave -> 16 rows x 64 cols.
// ---------------------------------------------------------------------------
__global__ __launch_bounds__(256)
void wmma_gemm_kernel(const float* __restrict__ A,
                      const unsigned short* __restrict__ Bb,
                      const float* __restrict__ bias,
                      float* __restrict__ out0, float* __restrict__ out1,
                      int M, int K, int do_relu) {
  const int lane = threadIdx.x & 31;
  const int wave = threadIdx.x >> 5;
  const int gw   = blockIdx.x * (blockDim.x >> 5) + wave;
  const int m0   = gw * 16;
  if (m0 >= M) return;                  // wave-uniform: EXEC stays all-1s

  v8f acc[4] = {};                      // 4 N-tiles of 16 -> 64 columns

  const int mrow  = m0 + (lane & 15);
  const int khalf = (lane >= 16) ? 8 : 0;   // A layout: lane halves 8 K apart

  for (int k0 = 0; k0 < K; k0 += 32) {
    // ---- A fragment: 16x32 bf16, ISA 7.12.2 layout, convert in-register ----
    v16u au;
#pragma unroll
    for (int v = 0; v < 8; ++v) {
      const int kb = k0 + ((v < 4) ? 0 : 16) + khalf + 2 * (v & 3);
      au[2 * v]     = f2bf(A[(size_t)mrow * K + kb]);
      au[2 * v + 1] = f2bf(A[(size_t)mrow * K + kb + 1]);
    }
    const v16bf afrag = __builtin_bit_cast(v16bf, au);

    const int ncol = lane & 15;         // B layout: lane holds column N=lane%16, K same as A
#pragma unroll
    for (int nt = 0; nt < 4; ++nt) {
      // lane's 16 bf16 of column nt*16+ncol, K in the same table order as A
      v16u bu;
#pragma unroll
      for (int s = 0; s < 16; ++s) {
        const int kk = k0 + ((s < 8) ? 0 : 16) + khalf + (s & 7);
        bu[s] = Bb[(size_t)kk * H_DIM + nt * 16 + ncol];
      }
      const v16bf bfrag = __builtin_bit_cast(v16bf, bu);
      acc[nt] = __builtin_amdgcn_wmma_f32_16x16x32_bf16(
          false, afrag, false, bfrag, (short)0, acc[nt], false, false);
    }
  }

  // ---- store: C/D layout VGPR r -> M = r (+8 for high lane half) ----
  const int col   = lane & 15;
  const int rbase = (lane >= 16) ? 8 : 0;
#pragma unroll
  for (int nt = 0; nt < 4; ++nt) {
#pragma unroll
    for (int r = 0; r < 8; ++r) {
      const int row = m0 + rbase + r;
      const int c   = nt * 16 + col;
      float v = acc[nt][r] + bias[c];
      if (do_relu) v = fmaxf(v, 0.0f);
      out0[(size_t)row * H_DIM + c] = v;
      if (out1) out1[(size_t)row * H_DIM + c] = v;
    }
  }
}

// ---------------------------------------------------------------------------
// Small helpers
// ---------------------------------------------------------------------------
__global__ void set_f32_kernel(float* p, int n, float v) {
  int i = blockIdx.x * blockDim.x + threadIdx.x;
  if (i < n) p[i] = v;
}

__global__ void deg_count_kernel(const int* __restrict__ dst, float* __restrict__ deg) {
  int e = blockIdx.x * blockDim.x + threadIdx.x;
  if (e < N_EDGES) atomicAdd(&deg[dst[e]], 1.0f);
}

__global__ void deg_to_dinv_kernel(float* __restrict__ d) {
  int i = blockIdx.x * blockDim.x + threadIdx.x;
  if (i < N_NODES) d[i] = rsqrtf(d[i]);   // deg >= 1 always (self loop)
}

// per-node attention dots: al = h . att_l, ar = h . att_r
__global__ __launch_bounds__(256)
void node_dots_kernel(const float* __restrict__ h, const float* __restrict__ wl,
                      const float* __restrict__ wr, float* __restrict__ al,
                      float* __restrict__ ar) {
  int i = blockIdx.x * blockDim.x + threadIdx.x;
  if (i >= N_NODES) return;
  const float4* hp = (const float4*)(h + (size_t)i * H_DIM);
  const float4* lp = (const float4*)wl;
  const float4* rp = (const float4*)wr;
  float sl = 0.0f, sr = 0.0f;
#pragma unroll
  for (int q = 0; q < 16; ++q) {
    float4 hv = hp[q], lv = lp[q], rv = rp[q];
    sl += hv.x * lv.x + hv.y * lv.y + hv.z * lv.z + hv.w * lv.w;
    sr += hv.x * rv.x + hv.y * rv.y + hv.z * rv.z + hv.w * rv.w;
  }
  al[i] = sl;
  ar[i] = sr;
}

// h_next = EPS*h0 + selfloop_alpha * h   (seeds the scatter accumulator)
__global__ __launch_bounds__(256)
void self_init_kernel(const float* __restrict__ h, const float* __restrict__ h0,
                      const float* __restrict__ al, const float* __restrict__ ar,
                      const float* __restrict__ dinv, float* __restrict__ hn) {
  int idx = blockIdx.x * blockDim.x + threadIdx.x;
  if (idx >= N_NODES * 16) return;
  const int i = idx >> 4, q = idx & 15;
  const float di = dinv[i];
  const float a  = tanhf(al[i] + ar[i]) * di * di;
  float4 hv  = ((const float4*)h)[(size_t)i * 16 + q];
  float4 h0v = ((const float4*)h0)[(size_t)i * 16 + q];
  float4 o;
  o.x = EPS_F * h0v.x + a * hv.x;
  o.y = EPS_F * h0v.y + a * hv.y;
  o.z = EPS_F * h0v.z + a * hv.z;
  o.w = EPS_F * h0v.w + a * hv.w;
  ((float4*)hn)[(size_t)i * 16 + q] = o;
}

__global__ void edge_alpha_kernel(const int* __restrict__ src, const int* __restrict__ dst,
                                  const float* __restrict__ al, const float* __restrict__ ar,
                                  const float* __restrict__ dinv, float* __restrict__ alpha) {
  int e = blockIdx.x * blockDim.x + threadIdx.x;
  if (e >= N_EDGES) return;
  const int s = src[e], d = dst[e];
  alpha[e] = tanhf(al[s] + ar[d]) * dinv[s] * dinv[d];
}

// scatter-add of h[src]*alpha into h_next[dst]; L2-resident atomics
__global__ __launch_bounds__(256)
void edge_scatter_kernel(const int* __restrict__ src, const int* __restrict__ dst,
                         const float* __restrict__ h, const float* __restrict__ alpha,
                         float* __restrict__ hn) {
  int idx = blockIdx.x * blockDim.x + threadIdx.x;
  if (idx >= N_EDGES * 16) return;
  const int e = idx >> 4, q = idx & 15;
  const int s = src[e], d = dst[e];
  const float a = alpha[e];
  float4 hv = ((const float4*)h)[(size_t)s * 16 + q];
  float* p = hn + (size_t)d * H_DIM + q * 4;
  atomicAdd(p + 0, hv.x * a);
  atomicAdd(p + 1, hv.y * a);
  atomicAdd(p + 2, hv.z * a);
  atomicAdd(p + 3, hv.w * a);
}

// global add pool: emb[batch[i]] += hout[i]
__global__ __launch_bounds__(256)
void pool_kernel(const float* __restrict__ hout, const int* __restrict__ batch,
                 float* __restrict__ emb) {
  int idx = blockIdx.x * blockDim.x + threadIdx.x;
  if (idx >= N_NODES * 16) return;
  const int i = idx >> 4, q = idx & 15;
  const int g = batch[i];
  float4 hv = ((const float4*)hout)[(size_t)i * 16 + q];
  float* p = emb + (size_t)g * H_DIM + q * 4;
  atomicAdd(p + 0, hv.x);
  atomicAdd(p + 1, hv.y);
  atomicAdd(p + 2, hv.z);
  atomicAdd(p + 3, hv.w);
}

static inline int cdiv(long long a, int b) { return (int)((a + b - 1) / b); }

extern "C" void kernel_launch(void* const* d_in, const int* in_sizes, int n_in,
                              void* d_out, int out_size, void* d_ws, size_t ws_size,
                              hipStream_t stream) {
  (void)in_sizes; (void)n_in; (void)out_size; (void)ws_size;

  const float* x     = (const float*)d_in[0];
  const int*   ei    = (const int*)d_in[1];
  const int*   src   = ei;
  const int*   dst   = ei + N_EDGES;
  const int*   batch = (const int*)d_in[2];
  const float* W_in  = (const float*)d_in[3];
  const float* b_in  = (const float*)d_in[4];
  const float* att_l = (const float*)d_in[5];
  const float* att_r = (const float*)d_in[6];
  const float* W_out = (const float*)d_in[7];
  const float* b_out = (const float*)d_in[8];

  float* out  = (float*)d_out;
  float* emb  = out;                         // [G,H]
  float* hout = out + N_GRAPHS * H_DIM;      // [N,H]

  float* ws = (float*)d_ws;
  size_t o = 0;
  unsigned short* wbin  = (unsigned short*)(ws + o); o += (F_INDIM * H_DIM) / 2; // 8192 bf16
  unsigned short* wbout = (unsigned short*)(ws + o); o += (H_DIM * H_DIM) / 2;   // 4096 bf16
  float* dinv  = ws + o; o += N_NODES;
  float* al    = ws + o; o += N_NODES;
  float* ar    = ws + o; o += N_NODES;
  float* alpha = ws + o; o += N_EDGES;       // 32B alignment holds for h buffers
  float* hA    = ws + o; o += (size_t)N_NODES * H_DIM;
  float* hB    = ws + o; o += (size_t)N_NODES * H_DIM;
  float* h0    = ws + o; o += (size_t)N_NODES * H_DIM;

  const dim3 blk(256);

  // --- weights -> bf16 once per launch (deterministic, graph-capture safe) ---
  cvt_bf16_kernel<<<cdiv(F_INDIM * H_DIM, 256), blk, 0, stream>>>(W_in, wbin, F_INDIM * H_DIM);
  cvt_bf16_kernel<<<cdiv(H_DIM * H_DIM, 256), blk, 0, stream>>>(W_out, wbout, H_DIM * H_DIM);

  // --- gcn_norm: deg (init 1 for self loop) -> rsqrt ---
  set_f32_kernel<<<cdiv(N_NODES, 256), blk, 0, stream>>>(dinv, N_NODES, 1.0f);
  deg_count_kernel<<<cdiv(N_EDGES, 256), blk, 0, stream>>>(dst, dinv);
  deg_to_dinv_kernel<<<cdiv(N_NODES, 256), blk, 0, stream>>>(dinv);

  // --- input projection: h = relu(x @ W_in + b_in); also h0 = h ---
  const int waves = N_NODES / 16;            // 3125 exactly
  wmma_gemm_kernel<<<cdiv(waves, 8), blk, 0, stream>>>(
      x, wbin, b_in, hA, h0, N_NODES, F_INDIM, 1);

  // --- 4 FAConv layers ---
  float* cur = hA;
  float* nxt = hB;
  for (int l = 0; l < N_LAYERS; ++l) {
    node_dots_kernel<<<cdiv(N_NODES, 256), blk, 0, stream>>>(
        cur, att_l + l * H_DIM, att_r + l * H_DIM, al, ar);
    self_init_kernel<<<cdiv((long long)N_NODES * 16, 256), blk, 0, stream>>>(
        cur, h0, al, ar, dinv, nxt);
    edge_alpha_kernel<<<cdiv(N_EDGES, 256), blk, 0, stream>>>(
        src, dst, al, ar, dinv, alpha);
    edge_scatter_kernel<<<cdiv((long long)N_EDGES * 16, 256), blk, 0, stream>>>(
        src, dst, cur, alpha, nxt);
    float* t = cur; cur = nxt; nxt = t;
  }

  // --- output projection: hout = h @ W_out + b_out ---
  wmma_gemm_kernel<<<cdiv(waves, 8), blk, 0, stream>>>(
      cur, wbout, b_out, hout, nullptr, N_NODES, H_DIM, 0);

  // --- global add pool ---
  set_f32_kernel<<<cdiv(N_GRAPHS * H_DIM, 256), blk, 0, stream>>>(
      emb, N_GRAPHS * H_DIM, 0.0f);
  pool_kernel<<<cdiv((long long)N_NODES * 16, 256), blk, 0, stream>>>(hout, batch, emb);
}
